// DiTAttention_20564303413793
// MI455X (gfx1250) — matrix operand order repaired and checked by reading, not compiled, not run on hardware
//
#include <hip/hip_runtime.h>
#include <hip/hip_bf16.h>

typedef _Float16 half_t;
typedef __attribute__((ext_vector_type(16))) _Float16 v16h;
typedef __attribute__((ext_vector_type(8)))  _Float16 v8h;
typedef __attribute__((ext_vector_type(4)))  _Float16 v4h;
typedef __attribute__((ext_vector_type(8)))  float    v8f;

typedef __attribute__((ext_vector_type(4))) unsigned int u32x4;
typedef __attribute__((ext_vector_type(8))) int          i32x8;
typedef __attribute__((ext_vector_type(4))) int          i32x4;

#define B_    2
#define SEQ   2048
#define DIM   1024
#define HEADS 16
#define HD    64
#define NT    (B_*SEQ)      // 4096 tokens
#define QKVN  (3*DIM)       // 3072

#define HAVE_TDM (__has_builtin(__builtin_amdgcn_tensor_load_to_lds) && \
                  __has_builtin(__builtin_amdgcn_s_wait_tensorcnt))

static __device__ __forceinline__ v16h cat16(v8h lo, v8h hi) {
  return __builtin_shufflevector(lo, hi, 0,1,2,3,4,5,6,7,8,9,10,11,12,13,14,15);
}
static __device__ __forceinline__ v8f wmma_f16(v16h a, v16h b, v8f c) {
  // v_wmma_f32_16x16x32_f16: D = A(16x32 f16) * B(32x16 f16) + C(16x16 f32)
  return __builtin_amdgcn_wmma_f32_16x16x32_f16(false, a, false, b, (short)0, c, false, false);
}

#if HAVE_TDM
// LDS byte offset of a __shared__ object: low 32 bits of the generic address
// (LDS aperture keeps the wave-relative offset in addr[31:0]).
static __device__ __forceinline__ unsigned lds_off(const void* p) {
  return (unsigned)(unsigned long long)(size_t)p;
}
// Issue a TDM 2-D tile load (f16 elements): tile_h rows x tile_w elems, row
// pitch stride_elems, from gaddr into LDS at lds_byte_off (rows packed
// contiguously: row y starts at lds_byte_off + y*tile_w*2).
// D# packing per CDNA5 ISA.
static __device__ __forceinline__ void tdm_load_2d(
    unsigned lds_byte_off, const void* gaddr,
    unsigned tile_w, unsigned tile_h, unsigned stride_elems) {
  unsigned long long ga = (unsigned long long)(size_t)gaddr;
  u32x4 g0;
  g0[0] = 1u;                                            // count=1 valid D#
  g0[1] = lds_byte_off;                                  // lds_addr
  g0[2] = (unsigned)(ga & 0xFFFFFFFFu);                  // global_addr[31:0]
  g0[3] = (unsigned)((ga >> 32) & 0x01FFFFFFu) | (2u << 30); // addr[56:32]|type=2
  i32x8 g1;
  g1[0] = (int)(1u << 16);                               // data_size=1 (2B)
  g1[1] = (int)((tile_w & 0xFFFFu) << 16);               // tensor_dim0[15:0]
  g1[2] = (int)(((tile_w >> 16) & 0xFFFFu) |
                ((tile_h & 0xFFFFu) << 16));             // dim0 hi | dim1 lo
  g1[3] = (int)(((tile_h >> 16) & 0xFFFFu) |
                ((tile_w & 0xFFFFu) << 16));             // dim1 hi | tile_dim0
  g1[4] = (int)(tile_h & 0xFFFFu);                       // tile_dim1, tile_dim2=0
  g1[5] = (int)stride_elems;                             // tensor_dim0_stride
  g1[6] = 0;
  g1[7] = 0;
  i32x4 z4 = {};
#if __clang_major__ >= 23
  i32x8 z8 = {};
  __builtin_amdgcn_tensor_load_to_lds(g0, g1, z4, z4, z8, 0);
#else
  __builtin_amdgcn_tensor_load_to_lds(g0, g1, z4, z4, 0);
#endif
}
#endif  // HAVE_TDM

// ---------------------------------------------------------------------------
// Kernel 1: f32 -> f16 conversion (vectorized by 4)
// ---------------------------------------------------------------------------
__global__ void cvt_f32_to_f16(const float* __restrict__ in,
                               half_t* __restrict__ out, int n4) {
  int i = blockIdx.x * 256 + threadIdx.x;
  if (i < n4) {
    float4 v = ((const float4*)in)[i];
    v4h o;
    o[0] = (half_t)v.x; o[1] = (half_t)v.y;
    o[2] = (half_t)v.z; o[3] = (half_t)v.w;
    ((v4h*)out)[i] = o;
  }
}

// ---------------------------------------------------------------------------
// Kernel 2/5: WMMA GEMM.  C[MxN] f32 = A[MxK] f16 * B[KxN] f16 + bias[N]
// Block tile 128x128, 8 waves in 2(M) x 4(N), each wave 4x2 16x16 tiles.
// Double-buffered LDS; A tile moved by the Tensor Data Mover (wave 0 issues
// one tensor_load_to_lds per k-step, TENSORcnt-waited before the barrier),
// B tile staged through registers with an n-major transpose.
// ---------------------------------------------------------------------------
__global__ __launch_bounds__(256) void gemm_f16_wmma(
    const half_t* __restrict__ A, const half_t* __restrict__ Bm,
    const float* __restrict__ bias, float* __restrict__ C,
    int M, int N, int K) {
  __shared__ __align__(16) half_t As[2][128 * 32];
  __shared__ __align__(16) half_t Bst[2][128 * 32];
  const int tid  = threadIdx.x;
  const int lane = tid & 31;
  const int wave = tid >> 5;
  const int wm = wave & 1, wn = wave >> 1;
  const int r = lane & 15, hi = lane >> 4;
  const int m0 = blockIdx.x * 128, n0 = blockIdx.y * 128;

  v8f acc[4][2] = {};
  const int bkr = tid >> 3, bcs = tid & 7;   // B tile: 32 k-rows x 8 segs of 16
  const half_t* Ablk = A + (size_t)m0 * K;
  const half_t* bptr = Bm + (size_t)bkr * N + n0 + bcs * 16;

  v8h rb0, rb1;                              // staged B regs
#if !HAVE_TDM
  const int arow = tid >> 1, aseg = tid & 1;
  const half_t* aptr = Ablk + (size_t)arow * K + aseg * 16;
  uint4 ra0, ra1;
#endif

  // prologue: fetch k-tile 0 into buffer 0
#if HAVE_TDM
  if (wave == 0) tdm_load_2d(lds_off(&As[0][0]), Ablk, 32, 128, K);
#else
  ra0 = *(const uint4*)aptr;
  ra1 = *(const uint4*)(aptr + 8);
  *(uint4*)&As[0][arow * 32 + aseg * 16]     = ra0;
  *(uint4*)&As[0][arow * 32 + aseg * 16 + 8] = ra1;
#endif
  rb0 = *(const v8h*)bptr;
  rb1 = *(const v8h*)(bptr + 8);
  #pragma unroll
  for (int t = 0; t < 8; t++) Bst[0][(bcs * 16 + t) * 32 + bkr]     = rb0[t];
  #pragma unroll
  for (int t = 0; t < 8; t++) Bst[0][(bcs * 16 + 8 + t) * 32 + bkr] = rb1[t];
#if HAVE_TDM
  __builtin_amdgcn_s_wait_tensorcnt(0);
#endif
  __syncthreads();

  int buf = 0;
  for (int k0 = 0; k0 < K; k0 += 32) {
    const bool has_next = (k0 + 32) < K;
    const int nb = buf ^ 1;
    if (has_next) { // issue next-tile data movement early (overlaps WMMAs)
#if HAVE_TDM
      if (wave == 0) tdm_load_2d(lds_off(&As[nb][0]), Ablk + k0 + 32, 32, 128, K);
#else
      const half_t* an = aptr + k0 + 32;
      ra0 = *(const uint4*)an;
      ra1 = *(const uint4*)(an + 8);
#endif
      const half_t* bn = bptr + (size_t)(k0 + 32) * N;
      rb0 = *(const v8h*)bn;
      rb1 = *(const v8h*)(bn + 8);
      if (k0 + 64 < K) __builtin_prefetch(bn + (size_t)32 * N, 0, 1);
    }

    // A frags: lane holds row M=r; halves j -> k = (j/8)*16 + hi*8 + j%8
    v16h af[4];
    #pragma unroll
    for (int i = 0; i < 4; i++) {
      const half_t* p = &As[buf][(wm * 64 + i * 16 + r) * 32];
      af[i] = cat16(*(const v8h*)(p + hi * 8), *(const v8h*)(p + 16 + hi * 8));
    }
    // B frags: lane holds col N=r; halves m -> k = (m/8)*16 + hi*8 + m%8
    v16h bf[2];
    #pragma unroll
    for (int j = 0; j < 2; j++) {
      const half_t* p = &Bst[buf][(wn * 32 + j * 16 + r) * 32 + hi * 8];
      bf[j] = cat16(*(const v8h*)p, *(const v8h*)(p + 16));
    }
    #pragma unroll
    for (int i = 0; i < 4; i++)
      #pragma unroll
      for (int j = 0; j < 2; j++)
        acc[i][j] = wmma_f16(af[i], bf[j], acc[i][j]);

    if (has_next) { // stage next B tile (transposed) into the other buffer
#if !HAVE_TDM
      *(uint4*)&As[nb][arow * 32 + aseg * 16]     = ra0;
      *(uint4*)&As[nb][arow * 32 + aseg * 16 + 8] = ra1;
#endif
      #pragma unroll
      for (int t = 0; t < 8; t++) Bst[nb][(bcs * 16 + t) * 32 + bkr]     = rb0[t];
      #pragma unroll
      for (int t = 0; t < 8; t++) Bst[nb][(bcs * 16 + 8 + t) * 32 + bkr] = rb1[t];
    }
#if HAVE_TDM
    __builtin_amdgcn_s_wait_tensorcnt(0);
#endif
    __syncthreads();
    buf = nb;
  }

  // Epilogue: D layout -> lane holds col n=r, rows m = e + 8*hi
  #pragma unroll
  for (int i = 0; i < 4; i++)
    #pragma unroll
    for (int j = 0; j < 2; j++) {
      int gn = n0 + wn * 32 + j * 16 + r;
      float bv = bias[gn];
      #pragma unroll
      for (int e = 0; e < 8; e++) {
        int gm = m0 + wm * 64 + i * 16 + e + 8 * hi;
        C[(size_t)gm * N + gn] = acc[i][j][e] + bv;
      }
    }
}

// ---------------------------------------------------------------------------
// Kernel 3: fused RMSNorm(q,k) + RoPE + head relayout; v relayout.
// ---------------------------------------------------------------------------
__global__ __launch_bounds__(256) void rmsnorm_rope_split(
    const float* __restrict__ qkv, const float* __restrict__ qs,
    const float* __restrict__ ksc, half_t* __restrict__ qh,
    half_t* __restrict__ kh, half_t* __restrict__ vh) {
  const int tok = blockIdx.x;
  const int b = tok >> 11, pos = tok & (SEQ - 1);
  const float* row = qkv + (size_t)tok * QKVN;
  const int tid = threadIdx.x, lane = tid & 31, wave = tid >> 5;

  float sq = 0.f, sk = 0.f;
  for (int i = tid; i < DIM; i += 256) {
    float qv = row[i], kv = row[DIM + i];
    sq += qv * qv; sk += kv * kv;
  }
  #pragma unroll
  for (int m = 16; m >= 1; m >>= 1) {
    sq += __shfl_xor(sq, m);
    sk += __shfl_xor(sk, m);
  }
  __shared__ float red[16];
  if (lane == 0) { red[wave] = sq; red[8 + wave] = sk; }
  __syncthreads();
  float tq = 0.f, tk = 0.f;
  #pragma unroll
  for (int w = 0; w < 8; w++) { tq += red[w]; tk += red[8 + w]; }
  const float rq = rsqrtf(tq * (1.f / DIM) + 1e-6f);
  const float rk = rsqrtf(tk * (1.f / DIM) + 1e-6f);

  for (int i = tid; i < DIM; i += 256) {
    int h = i >> 6, j = i & 63, jj = j & 31;
    int pidx = (j < 32) ? (i + 32) : (i - 32);
    float sgn = (j < 32) ? -1.f : 1.f;
    // inv_freq = 10000^(-jj/32) = exp(-ln(10000)/32 * jj)
    float inv = __expf(-0.2878231366f * (float)jj);
    float th = (float)pos * inv;
    float c = __cosf(th), s = __sinf(th);
    float qn = row[i] * rq * qs[i];
    float qp = row[pidx] * rq * qs[pidx];
    float kn = row[DIM + i] * rk * ksc[i];
    float kp = row[DIM + pidx] * rk * ksc[pidx];
    size_t o = ((size_t)(b * HEADS + h) * SEQ + pos) * HD + j;
    qh[o] = (half_t)(qn * c + sgn * qp * s);
    kh[o] = (half_t)(kn * c + sgn * kp * s);
    vh[o] = (half_t)row[2 * DIM + i];
  }
}

// ---------------------------------------------------------------------------
// Kernel 4: flash attention. grid = (b*h = 32, qblocks = 16), 8 waves,
// each wave owns 16 query rows; 64-key blocks, double-buffered K/V in LDS.
// K tile moved by the TDM; V staged transposed through registers.
// ---------------------------------------------------------------------------
__global__ __launch_bounds__(256) void flash_attn(
    const half_t* __restrict__ qh, const half_t* __restrict__ kh,
    const half_t* __restrict__ vh, half_t* __restrict__ ao) {
  __shared__ __align__(16) half_t Kl[2][64 * 64];    // [key][d]
  __shared__ __align__(16) half_t Vt[2][64 * 64];    // [d(hd)][key] transposed
  __shared__ __align__(16) half_t Pl[8 * 16 * 64];   // per-wave P tile
  const int tid = threadIdx.x, lane = tid & 31, wave = tid >> 5;
  const int r = lane & 15, hi = lane >> 4;
  const int bh = blockIdx.x;
  const int b = bh >> 4, h = bh & 15;
  const int qb = blockIdx.y * 128;
  const half_t* Q  = qh + (size_t)bh * SEQ * HD;
  const half_t* Kg = kh + (size_t)bh * SEQ * HD;
  const half_t* Vg = vh + (size_t)bh * SEQ * HD;

  // Q A-fragments (2 chunks of K=32 over d=64), 1/sqrt(64)=0.125 folded in
  const int qrow = qb + wave * 16 + r;
  v16h qa[2];
  #pragma unroll
  for (int c = 0; c < 2; c++) {
    const half_t* p = Q + (size_t)qrow * HD + c * 32;
    v8h lo = *(const v8h*)(p + hi * 8);
    v8h hv = *(const v8h*)(p + 16 + hi * 8);
    #pragma unroll
    for (int t = 0; t < 8; t++) { lo[t] *= (_Float16)0.125f; hv[t] *= (_Float16)0.125f; }
    qa[c] = cat16(lo, hv);
  }

  v8f o[4] = {};
  float mrow[8], lrow[8];
  #pragma unroll
  for (int i = 0; i < 8; i++) { mrow[i] = -1e30f; lrow[i] = 0.f; }

  const int ldr = tid >> 2, ldc = tid & 3;     // cooperative V tile load
  half_t* Pw = &Pl[wave * 16 * 64];

  v8h rv0, rv1;                                // staged V regs
#if !HAVE_TDM
  uint4 rk0, rk1;
#endif

  // prologue: fetch key-block 0 into buffer 0
#if HAVE_TDM
  if (wave == 0) tdm_load_2d(lds_off(&Kl[0][0]), Kg, 64, 64, 64);
#else
  {
    const half_t* s = Kg + (size_t)ldr * HD + ldc * 16;
    rk0 = *(const uint4*)s;
    rk1 = *(const uint4*)(s + 8);
    *(uint4*)&Kl[0][ldr * 64 + ldc * 16]     = rk0;
    *(uint4*)&Kl[0][ldr * 64 + ldc * 16 + 8] = rk1;
  }
#endif
  {
    const half_t* v = Vg + (size_t)ldr * HD + ldc * 16;
    rv0 = *(const v8h*)v;
    rv1 = *(const v8h*)(v + 8);
    #pragma unroll
    for (int t = 0; t < 8; t++) Vt[0][(ldc * 16 + t) * 64 + ldr]     = rv0[t];
    #pragma unroll
    for (int t = 0; t < 8; t++) Vt[0][(ldc * 16 + 8 + t) * 64 + ldr] = rv1[t];
  }
#if HAVE_TDM
  __builtin_amdgcn_s_wait_tensorcnt(0);
#endif
  __syncthreads();

  int buf = 0;
  for (int kb = 0; kb < SEQ; kb += 64) {
    const bool has_next = (kb + 64) < SEQ;
    const int nb = buf ^ 1;
    if (has_next) { // issue next key-block movement early
#if HAVE_TDM
      if (wave == 0)
        tdm_load_2d(lds_off(&Kl[nb][0]), Kg + (size_t)(kb + 64) * HD, 64, 64, 64);
#else
      const half_t* s = Kg + (size_t)(kb + 64 + ldr) * HD + ldc * 16;
      rk0 = *(const uint4*)s;
      rk1 = *(const uint4*)(s + 8);
#endif
      const half_t* v = Vg + (size_t)(kb + 64 + ldr) * HD + ldc * 16;
      rv0 = *(const v8h*)v;
      rv1 = *(const v8h*)(v + 8);
    }

    // S = (Q*scale) @ K^T : 4 key-tiles of 16, d = 64 in 2 chunks of 32
    v8f zero8 = {};
    v8f s4[4];
    #pragma unroll
    for (int t = 0; t < 4; t++) {
      s4[t] = zero8;
      #pragma unroll
      for (int c = 0; c < 2; c++) {
        const half_t* p = &Kl[buf][(t * 16 + r) * 64 + c * 32 + hi * 8];
        v16h bfr = cat16(*(const v8h*)p, *(const v8h*)(p + 16));
        s4[t] = wmma_f16(qa[c], bfr, s4[t]);
      }
    }

    // online softmax: row (i + 8*hi) lives in register i across 16 lanes
    #pragma unroll
    for (int i = 0; i < 8; i++) {
      float mx = fmaxf(fmaxf(s4[0][i], s4[1][i]), fmaxf(s4[2][i], s4[3][i]));
      #pragma unroll
      for (int m = 8; m >= 1; m >>= 1) mx = fmaxf(mx, __shfl_xor(mx, m));
      float mnew  = fmaxf(mrow[i], mx);
      float alpha = __expf(mrow[i] - mnew);
      float rs = 0.f;
      #pragma unroll
      for (int t = 0; t < 4; t++) {
        float e = __expf(s4[t][i] - mnew);
        s4[t][i] = e; rs += e;
      }
      #pragma unroll
      for (int m = 8; m >= 1; m >>= 1) rs += __shfl_xor(rs, m);
      lrow[i] = lrow[i] * alpha + rs;
      mrow[i] = mnew;
      #pragma unroll
      for (int t = 0; t < 4; t++) o[t][i] *= alpha;
    }

    // P (D-layout) -> per-wave LDS -> reload as A-fragments
    #pragma unroll
    for (int t = 0; t < 4; t++)
      #pragma unroll
      for (int i = 0; i < 8; i++)
        Pw[(i + 8 * hi) * 64 + t * 16 + r] = (half_t)s4[t][i];
    asm volatile("s_wait_dscnt 0" ::: "memory");  // per-wave LDS RAW guard

    // O += P @ V  (V staged transposed so B-frags are contiguous)
    #pragma unroll
    for (int c = 0; c < 2; c++) {
      const half_t* pp = &Pw[(size_t)r * 64 + c * 32];
      v16h pa = cat16(*(const v8h*)(pp + hi * 8), *(const v8h*)(pp + 16 + hi * 8));
      #pragma unroll
      for (int t = 0; t < 4; t++) {
        const half_t* vp = &Vt[buf][(t * 16 + r) * 64 + c * 32 + hi * 8];
        v16h vb = cat16(*(const v8h*)vp, *(const v8h*)(vp + 16));
        o[t] = wmma_f16(pa, vb, o[t]);
      }
    }

    if (has_next) { // stage next key-block into the other buffer
#if !HAVE_TDM
      *(uint4*)&Kl[nb][ldr * 64 + ldc * 16]     = rk0;
      *(uint4*)&Kl[nb][ldr * 64 + ldc * 16 + 8] = rk1;
#endif
      #pragma unroll
      for (int t = 0; t < 8; t++) Vt[nb][(ldc * 16 + t) * 64 + ldr]     = rv0[t];
      #pragma unroll
      for (int t = 0; t < 8; t++) Vt[nb][(ldc * 16 + 8 + t) * 64 + ldr] = rv1[t];
    }
#if HAVE_TDM
    __builtin_amdgcn_s_wait_tensorcnt(0);
#endif
    __syncthreads();
    buf = nb;
  }

  // normalize and scatter to [b][n][h*64 + hd] (f16, feeds final GEMM)
  #pragma unroll
  for (int t = 0; t < 4; t++)
    #pragma unroll
    for (int i = 0; i < 8; i++) {
      int q = qb + wave * 16 + i + 8 * hi;
      float val = o[t][i] / lrow[i];
      ao[((size_t)b * SEQ + q) * DIM + h * HD + t * 16 + r] = (half_t)val;
    }
}

// ---------------------------------------------------------------------------
extern "C" void kernel_launch(void* const* d_in, const int* in_sizes, int n_in,
                              void* d_out, int out_size, void* d_ws, size_t ws_size,
                              hipStream_t stream) {
  (void)in_sizes; (void)n_in; (void)out_size; (void)ws_size;
  const float* input   = (const float*)d_in[0];
  const float* w_qkv   = (const float*)d_in[1];
  const float* b_qkv   = (const float*)d_in[2];
  const float* q_scale = (const float*)d_in[3];
  const float* k_scale = (const float*)d_in[4];
  const float* w_out   = (const float*)d_in[5];
  const float* b_out   = (const float*)d_in[6];
  float* out = (float*)d_out;

  char* ws = (char*)d_ws;
  const size_t MB = 1024 * 1024;
  half_t* x16  = (half_t*)(ws);              // 4096x1024 f16  (8 MB)
  half_t* wq16 = (half_t*)(ws +  8 * MB);    // 1024x3072 f16  (6 MB)
  half_t* wo16 = (half_t*)(ws + 14 * MB);    // 1024x1024 f16  (2 MB)
  float*  qkvf = (float*) (ws + 16 * MB);    // 4096x3072 f32  (48 MB)
  half_t* qhp  = (half_t*)(ws + 64 * MB);    // [b][h][n][64]  (8 MB)
  half_t* khp  = (half_t*)(ws + 72 * MB);    // (8 MB)
  half_t* vhp  = (half_t*)(ws + 80 * MB);    // (8 MB)
  half_t* aop  = (half_t*)(ws + 88 * MB);    // 4096x1024 f16  (8 MB)

  cvt_f32_to_f16<<<4096, 256, 0, stream>>>(input, x16, NT * DIM / 4);
  cvt_f32_to_f16<<<3072, 256, 0, stream>>>(w_qkv, wq16, DIM * QKVN / 4);
  cvt_f32_to_f16<<<1024, 256, 0, stream>>>(w_out, wo16, DIM * DIM / 4);

  gemm_f16_wmma<<<dim3(32, 24), 256, 0, stream>>>(x16, wq16, b_qkv, qkvf,
                                                  NT, QKVN, DIM);
  rmsnorm_rope_split<<<NT, 256, 0, stream>>>(qkvf, q_scale, k_scale,
                                             qhp, khp, vhp);
  flash_attn<<<dim3(32, 16), 256, 0, stream>>>(qhp, khp, vhp, aop);
  gemm_f16_wmma<<<dim3(32, 8), 256, 0, stream>>>(aop, wo16, b_out, out,
                                                 NT, DIM, DIM);
}
